// DistanceWeightedSampling_58411555226013
// MI455X (gfx1250) — matrix operand order repaired and checked by reading, not compiled, not run on hardware
//
#include <hip/hip_runtime.h>
#include <hip/hip_bf16.h>
#include <stdint.h>

// ---------------------------------------------------------------------------
// DistanceWeightedSampling on MI455X (gfx1250, wave32)
//
// sim = x x^T fused with distance-weighted categorical sampling of 7
// negatives per row. GEMM: v_wmma_f32_16x16x32_bf16 (bf16 in, f32 accum),
// K=128 = 4 chained WMMAs per 16x16 tile, B fragments clause-loaded, next
// tile prefetched (global_prefetch_b8). Sampling uses the exponential-race
// identity in base 2:  argmin_j (-log2 u_j) / w_j  ~  Categorical(w),
// with 1/w = exp2(72 - log2w) precomputed per element -> exactly one raw
// v_log_f32 per (element, sample) and raw v_exp_f32 per element (inputs are
// provably normal-range, so no libm fixup code). sqrt eliminated (runs on
// d^2). Per-row argmin: single-u32 key with lane id embedded in low 4 bits,
// reduced by 4x (ds_swizzle + v_max_u32), winner committed via ds_max_u64.
// ---------------------------------------------------------------------------

#define N_PTS   8192
#define D_DIM   128
#define KM1     7            // k-1 negatives per anchor
#define TILE    16
#define NTILES  (N_PTS / TILE)      // 512
#define WPB     8                    // waves per block

// flat float offsets into d_out (outputs concatenated in return order)
#define O1 57344LL                       // x[a_idx]   [57344,128]
#define O2 (O1 + 7340032LL)              // x[p_idx]
#define O3 (O2 + 7340032LL)              // x[n_idx]
#define O4 (O3 + 7340032LL)              // x          [8192,128]

typedef __attribute__((ext_vector_type(16))) __bf16 v16bf;
typedef __attribute__((ext_vector_type(8)))  __bf16 v8bf;
typedef __attribute__((ext_vector_type(8)))  float  v8f;

__device__ __forceinline__ unsigned avalanche(unsigned h) {
    h ^= h >> 16; h *= 0x7FEB352Du;
    h ^= h >> 15; h *= 0x846CA68Bu;
    h ^= h >> 16;
    return h;
}

// ---------------------------- Pass A: f32 -> bf16 ---------------------------
__global__ __launch_bounds__(256) void cvt_bf16_kernel(const float* __restrict__ x,
                                                       __bf16* __restrict__ xb) {
    int i = (blockIdx.x * 256 + threadIdx.x) * 4;
#pragma unroll
    for (int t = 0; t < 4; ++t) xb[i + t] = (__bf16)x[i + t];
}

// ------------------- Pass B: WMMA GEMM + fused categorical sampling ----------
// One block owns a 16-row tile of sim against all 8192 columns.
__global__ __launch_bounds__(256) void sample_kernel(const __bf16* __restrict__ xb,
                                                     int* __restrict__ nidx) {
    __shared__ unsigned long long best[TILE][KM1];   // (key<<32)|col per (row,s)

    const int tid     = threadIdx.x;
    const int lane    = tid & 31;
    const int wave    = tid >> 5;
    const int laneLo  = lane & 15;
    const int laneHi  = lane >> 4;
    const int rowBase = blockIdx.x * TILE;

    if (tid < TILE * KM1) {
        int r = tid / KM1, s = tid % KM1;
        // uniform fallback index for degenerate rows (key = 0 loses to anything)
        unsigned fb = avalanche((rowBase + r) * 0x9E3779B1u + s * 0x85EBCA77u
                                + 0x51F15EEDu) & (N_PTS - 1);
        best[r][s] = (unsigned long long)fb;
    }
    __syncthreads();

    // A fragments: 16x128 row tile, contiguous K layout (matches B):
    //   lanes 0-15  : M=lane,    K c0..c0+15, c0 = kb
    //   lanes 16-31 : M=lane-16, same with c0 = kb+16
    const __bf16* arowp = xb + (rowBase + laneLo) * D_DIM;
    v16bf A[4];
#pragma unroll
    for (int kc = 0; kc < 4; ++kc) {
        int c0 = kc * 32 + laneHi * 16;
        A[kc] = *(const v16bf*)(arowp + c0);
    }

    for (int ct = wave; ct < NTILES; ct += WPB) {
        const int col = ct * TILE + laneLo;
        // B layout (32x16, K-major): lane = N, lanes 0-15 hold K 0..15,
        // lanes 16-31 hold K 16..31 -> 16 contiguous bf16 per lane.
        const __bf16* bcolp = xb + col * D_DIM + laneHi * 16;

        // clause-load all four B fragments, then run WMMAs back-to-back
        v16bf B[4];
#pragma unroll
        for (int kc = 0; kc < 4; ++kc) B[kc] = *(const v16bf*)(bcolp + kc * 32);

        // prefetch next column tile's B rows while we compute
        int nct = ct + WPB;
        if (nct < NTILES)
            __builtin_prefetch(xb + (nct * TILE + laneLo) * D_DIM + laneHi * 16, 0, 1);

        v8f acc = {};
#pragma unroll
        for (int kc = 0; kc < 4; ++kc)
            acc = __builtin_amdgcn_wmma_f32_16x16x32_bf16(
                false, A[kc], false, B[kc], (short)0, acc, false, false);

        // Epilogue on d^2 (no sqrt), all base-2 raw transcendentals:
        //   d2c = max(max(2-2 sim, 0), 0.25)            (CUTOFF^2)
        //   q   = log2(w) = -63 log2(d2c) - 62.5 log2(max(1-0.25 d2c, 1e-8))
        //   valid = off-block && d2 < 1.96  (== dist < 1.4)
        //   inv = 2^(72 - q)   (q in [-0.5, 131.8] -> exponent in [-59.8, 72.5])
        // C/D layout: VGPR r, lanes 0-15 -> row rowBase+r; lanes 16-31 -> +8.
        float inv[8];
        unsigned vmask = 0;
#pragma unroll
        for (int r = 0; r < 8; ++r) {
            int   row = rowBase + r + 8 * laneHi;
            float d2  = fmaxf(2.0f - 2.0f * acc[r], 0.0f);
            float d2c = fmaxf(d2, 0.25f);
            float om  = fmaxf(1.0f - 0.25f * d2c, 1e-8f);
            float q   = -63.0f * __builtin_amdgcn_logf(d2c)
                      - 62.5f * __builtin_amdgcn_logf(om);
            inv[r]    = __builtin_amdgcn_exp2f(72.0f - q);
            if (((row >> 3) != (col >> 3)) && (d2 < 1.96f)) vmask |= 1u << r;
        }

#pragma unroll
        for (int r = 0; r < 8; ++r) {
            int row = rowBase + r + 8 * laneHi;
            unsigned h0 = (unsigned)row * 0x9E3779B1u
                        + (unsigned)col * 0xC2B2AE3Du + 0x27220A95u;
            bool v = (vmask >> r) & 1u;
            for (int s = 0; s < KM1; ++s) {
                unsigned h = avalanche(h0 + (unsigned)s * 0x85EBCA77u);
                // u in [2^-24, 1-2^-24]; floor -log2(u) away from 0
                float u    = (float)((h >> 8) | 1u) * (1.0f / 16777216.0f);
                float e2   = fmaxf(-__builtin_amdgcn_logf(u), 1e-9f); // ~Exp/ln2
                float cand = e2 * inv[r];                 // race value, > 0
                // monotone min-key: smaller cand -> larger key; embed laneLo
                unsigned key = v ? (((0x7FFFFFFFu - __float_as_uint(cand))
                                     & 0xFFFFFFF0u) | (unsigned)laneLo)
                                 : 0u;
                // butterfly max across the 16 lanes sharing this row (u32 only)
#pragma unroll
                for (int m = 1; m <= 8; m <<= 1) {
                    unsigned okey = __shfl_xor(key, m, 32);
                    key = okey > key ? okey : key;
                }
                if (laneLo == 0) {
                    unsigned wcol = (unsigned)(ct * TILE) + (key & 15u);
                    atomicMax(&best[r + 8 * laneHi][s],
                              ((unsigned long long)key << 32) |
                               (unsigned long long)wcol);
                }
            }
        }
    }
    __syncthreads();

    if (tid < TILE * KM1) {
        int r = tid / KM1, s = tid % KM1;
        nidx[(rowBase + r) * KM1 + s] = (int)(best[r][s] & 0xFFFFFFFFull);
    }
}

// ---------------------- Pass C: gathers (pure bandwidth) ---------------------
__global__ __launch_bounds__(256) void gather_kernel(const float* __restrict__ x,
                                                     const int* __restrict__ nidx,
                                                     float* __restrict__ out) {
    long long gid = (long long)blockIdx.x * 256 + threadIdx.x; // [0, 57344*32)
    int m  = (int)(gid >> 5);
    int c4 = (int)(gid & 31);
    int i    = m / KM1;
    int t    = m - i * KM1;
    int base = (i >> 3) << 3;                 // block start (k=8)
    int off  = i - base;
    int p    = base + t + (t >= off ? 1 : 0); // ascending block members != i
    int nj   = nidx[m];

    const float4* xi = (const float4*)(x + (long long)i  * D_DIM);
    const float4* xp = (const float4*)(x + (long long)p  * D_DIM);
    const float4* xn = (const float4*)(x + (long long)nj * D_DIM);

    if (c4 == 0) out[m] = (float)i;           // a_idx
    ((float4*)(out + O1))[(long long)m * 32 + c4] = xi[c4];
    ((float4*)(out + O2))[(long long)m * 32 + c4] = xp[c4];
    ((float4*)(out + O3))[(long long)m * 32 + c4] = xn[c4];
}

__global__ __launch_bounds__(256) void copyx_kernel(const float* __restrict__ x,
                                                    float* __restrict__ out) {
    long long gid = (long long)blockIdx.x * 256 + threadIdx.x;
    ((float4*)(out + O4))[gid] = ((const float4*)x)[gid];
}

// ----------------------------------------------------------------------------
extern "C" void kernel_launch(void* const* d_in, const int* in_sizes, int n_in,
                              void* d_out, int out_size, void* d_ws, size_t ws_size,
                              hipStream_t stream) {
    (void)in_sizes; (void)n_in; (void)out_size; (void)ws_size;

    const float* x   = (const float*)d_in[0];
    float*       out = (float*)d_out;

    // workspace: [0, 2MB) bf16 copy of x; then n_idx ints
    __bf16* xb   = (__bf16*)d_ws;
    int*    nidx = (int*)((char*)d_ws + (size_t)N_PTS * D_DIM * 2);

    cvt_bf16_kernel<<<(N_PTS * D_DIM) / (256 * 4), 256, 0, stream>>>(x, xb);
    sample_kernel<<<NTILES, 256, 0, stream>>>(xb, nidx);
    gather_kernel<<<(N_PTS * KM1 * 32) / 256, 256, 0, stream>>>(x, nidx, out);
    copyx_kernel<<<(N_PTS * D_DIM / 4) / 256, 256, 0, stream>>>(x, out);
}
